// MetaGNN_24773371363902
// MI455X (gfx1250) — matrix operand order repaired and checked by reading, not compiled, not run on hardware
//
#include <hip/hip_runtime.h>
#include <stdint.h>

// ---------------------------------------------------------------------------
// MetaGNN forward on MI455X (gfx1250).
// - All big row-parallel MLPs via v_wmma_f32_16x16x32_bf16 (f32 accum).
// - Gather/concat staging into LDS with GLOBAL_LOAD_ASYNC_TO_LDS_B128 in
//   SADDR form (scalar ws base + 32-bit per-lane offset, offset:16 for the
//   second chunk), double-buffered and pipelined on ASYNCcnt.
// - K1 is a template parameter (256/384/512) so the K-loop fully unrolls and
//   the per-stage segment select folds at compile time (no alloca, no
//   scratch traffic in the hot loop).
// - scatter_mean numerator fused into the message-MLP epilogue via
//   global_atomic_add_f32.
// ---------------------------------------------------------------------------

typedef __attribute__((ext_vector_type(16))) __bf16          v16bf;
typedef __attribute__((ext_vector_type(8)))  float           v8f;
typedef __attribute__((ext_vector_type(8)))  unsigned short  v8s;
typedef long long i64;

#define ASYNC_STAGING 1

static __device__ __forceinline__ unsigned short f2bf(float f) {
  union { float f; unsigned u; } v; v.f = f;
  unsigned r = v.u + 0x7FFFu + ((v.u >> 16) & 1u);   // round-to-nearest-even
  return (unsigned short)(r >> 16);
}

union FragBF { v8s s[2]; v16bf v; };

// low 32 bits of a generic pointer to __shared__ == wave-relative LDS offset
static __device__ __forceinline__ unsigned lds_lo(const void* p) {
  return (unsigned)(unsigned long long)p;
}

// 32B/lane async copy global->LDS as two b128 ops (ASYNCcnt += 2 per wave).
// INST_OFFSET applies to BOTH the LDS destination and the global address,
// so offset:16 stages the second 16B chunk with no extra address math.
static __device__ __forceinline__ void async_pair(unsigned lds, unsigned voff,
                                                  const void* sbase) {
#if ASYNC_STAGING
  asm volatile("global_load_async_to_lds_b128 %0, %1, %2\n\t"
               "global_load_async_to_lds_b128 %0, %1, %2 offset:16"
               :: "v"(lds), "v"(voff), "s"(sbase)
               : "memory");
#else
  (void)lds; (void)voff; (void)sbase;
#endif
}

#define TILE_M 128
#define LDA    40     // halfs per row in staging tiles (16B-aligned, bank skew)
#define LDH    136    // halfs per row in hidden tile

// Fused 2-layer MLP:  out = (relu(concat(segs) @ W1 + b1)) @ W2 + b2
// Segments are bf16, width 128, addressed as 32-bit byte offsets into wsBase.
// mode per 4-bit nibble of segModes:
//   0 = gather idxA[row], 1 = gather idxB[row], 2 = direct row, 3 = broadcast
template <int K1>
__global__ __launch_bounds__(256, 1)
void fused_mlp2(
    i64 M,
    const char* __restrict__ wsBase,
    unsigned w1Off, const float* __restrict__ b1,
    unsigned w2Off, const float* __restrict__ b2,
    unsigned o0, unsigned o1, unsigned o2, unsigned o3,
    int segModes,
    const i64* __restrict__ idxA, const i64* __restrict__ idxB,
    float* __restrict__ outF, unsigned short* __restrict__ outBf,
    const i64* __restrict__ scatIdx, float* __restrict__ scatOut)
{
  __shared__ unsigned short sA0[TILE_M * LDA];
  __shared__ unsigned short sA1[TILE_M * LDA];
  __shared__ unsigned short sB0[TILE_M * LDA];
  __shared__ unsigned short sB1[TILE_M * LDA];
  __shared__ unsigned short sH[TILE_M * LDH];
  __shared__ float sBias1[128];
  __shared__ float sBias2[128];

  const int tid = threadIdx.x;
  if (tid < 128) { sBias1[tid] = b1[tid]; sBias2[tid] = b2[tid]; }

  const int lane = tid & 31;
  const int wave = tid >> 5;
  const int hi   = lane >> 4;
  const int mr   = lane & 15;
  const int m0   = wave * 16;        // 16-row strip per wave

  // staging coordinates: 2 threads per row, 16 halfs each
  const int  srow = tid >> 1;
  const int  soff = (tid & 1) * 16;
  const i64  grow = (i64)blockIdx.x * TILE_M + srow;
  const bool inB  = grow < M;

  // per-thread u32 row-base byte offsets into wsBase (gathers hoisted here)
  auto rowOff = [&](unsigned segOff, int mode) -> unsigned {
    i64 r = 0;
    if (inB) {
      if      (mode == 0) r = idxA[grow];
      else if (mode == 1) r = idxB[grow];
      else if (mode == 2) r = grow;
    }
    return segOff + (unsigned)(r * 256);   // 128 halfs * 2B per row
  };
  const unsigned ro0 = rowOff(o0, (segModes >> 0)  & 15);
  const unsigned ro1 = rowOff(o1, (segModes >> 4)  & 15);
  const unsigned ro2 = rowOff(o2, (segModes >> 8)  & 15);
  const unsigned ro3 = rowOff(o3, (segModes >> 12) & 15);

  // stage one 128x32 A tile + 128x32 B tile (layer 1): 4 async ops / wave.
  // ks is compile-time after unrolling -> seg select folds to a register.
  auto issueStage1 = [&](int ks, unsigned short* bufA, unsigned short* bufB) {
    const int c = ks * 32 + soff;
    const int s = ks / 4;                  // = c >> 7, constant per stage
    const unsigned ro = (s == 0) ? ro0 : (s == 1) ? ro1 : (s == 2) ? ro2 : ro3;
    const unsigned va = ro + (unsigned)((c & 127) * 2);
    const unsigned wv = w1Off + (unsigned)(srow * K1 + c) * 2u;
#if ASYNC_STAGING
    async_pair(lds_lo(&bufA[srow * LDA + soff]), va, wsBase);
    async_pair(lds_lo(&bufB[srow * LDA + soff]), wv, wsBase);
#else
    const v8s* ap = (const v8s*)(wsBase + va);
    const v8s* wp = (const v8s*)(wsBase + wv);
    v8s a0 = ap[0], a1 = ap[1], w0 = wp[0], w1 = wp[1];
    *(v8s*)&bufA[srow * LDA + soff]     = a0;
    *(v8s*)&bufA[srow * LDA + soff + 8] = a1;
    *(v8s*)&bufB[srow * LDA + soff]     = w0;
    *(v8s*)&bufB[srow * LDA + soff + 8] = w1;
#endif
  };

  // stage one 128x32 W2 tile (layer 2): 2 async ops / wave
  auto issueStage2 = [&](int ks, unsigned short* bufB) {
    const unsigned wv = w2Off + (unsigned)(srow * 128 + ks * 32 + soff) * 2u;
#if ASYNC_STAGING
    async_pair(lds_lo(&bufB[srow * LDA + soff]), wv, wsBase);
#else
    const v8s* wp = (const v8s*)(wsBase + wv);
    v8s w0 = wp[0], w1 = wp[1];
    *(v8s*)&bufB[srow * LDA + soff]     = w0;
    *(v8s*)&bufB[srow * LDA + soff + 8] = w1;
#endif
  };

  v8f acc[8];
  #pragma unroll
  for (int t = 0; t < 8; ++t)
    for (int i = 0; i < 8; ++i) acc[t][i] = 0.f;

  // ---------------- layer 1: K1 -> 128 (bias + relu) ----------------
  constexpr int ns1 = K1 / 32;
  issueStage1(0, sA0, sB0);
  #pragma unroll
  for (int ks = 0; ks < ns1; ++ks) {
    unsigned short* cA = (ks & 1) ? sA1 : sA0;
    unsigned short* cB = (ks & 1) ? sB1 : sB0;
    if (ks + 1 < ns1) {
      issueStage1(ks + 1, (ks & 1) ? sA0 : sA1, (ks & 1) ? sB0 : sB1);
#if ASYNC_STAGING
      asm volatile("s_wait_asynccnt 0x4" ::: "memory");  // older stage retired
#endif
    } else {
#if ASYNC_STAGING
      asm volatile("s_wait_asynccnt 0x0" ::: "memory");
#endif
    }
    __syncthreads();   // staged tile visible to all waves

    FragBF A;
    A.s[0] = *(const v8s*)&cA[(m0 + mr) * LDA + hi * 8];
    A.s[1] = *(const v8s*)&cA[(m0 + mr) * LDA + 16 + hi * 8];
    #pragma unroll
    for (int t = 0; t < 8; ++t) {
      FragBF B;
      B.s[0] = *(const v8s*)&cB[(t * 16 + mr) * LDA + hi * 8];
      B.s[1] = *(const v8s*)&cB[(t * 16 + mr) * LDA + 16 + hi * 8];
      acc[t] = __builtin_amdgcn_wmma_f32_16x16x32_bf16(
          false, A.v, false, B.v, (short)0, acc[t], false, false);
    }
    __syncthreads();   // reads done before this buffer is re-staged
  }

  // bias + relu -> hidden tile (bf16 in LDS)
  #pragma unroll
  for (int t = 0; t < 8; ++t) {
    int n = t * 16 + mr;
    float bv = sBias1[n];
    #pragma unroll
    for (int r = 0; r < 8; ++r) {
      float v = acc[t][r] + bv;
      v = v > 0.f ? v : 0.f;
      sH[(m0 + r + 8 * hi) * LDH + n] = f2bf(v);
    }
  }
  #pragma unroll
  for (int t = 0; t < 8; ++t)
    for (int i = 0; i < 8; ++i) acc[t][i] = 0.f;

  // ---------------- layer 2: 128 -> 128 ----------------
  issueStage2(0, sB0);
  #pragma unroll
  for (int ks = 0; ks < 4; ++ks) {
    unsigned short* cB = (ks & 1) ? sB1 : sB0;
    if (ks + 1 < 4) {
      issueStage2(ks + 1, (ks & 1) ? sB0 : sB1);
#if ASYNC_STAGING
      asm volatile("s_wait_asynccnt 0x2" ::: "memory");
#endif
    } else {
#if ASYNC_STAGING
      asm volatile("s_wait_asynccnt 0x0" ::: "memory");
#endif
    }
    __syncthreads();   // W2 tile staged; first iter also publishes sH

    FragBF A;
    A.s[0] = *(const v8s*)&sH[(m0 + mr) * LDH + ks * 32 + hi * 8];
    A.s[1] = *(const v8s*)&sH[(m0 + mr) * LDH + ks * 32 + 16 + hi * 8];
    #pragma unroll
    for (int t = 0; t < 8; ++t) {
      FragBF B;
      B.s[0] = *(const v8s*)&cB[(t * 16 + mr) * LDA + hi * 8];
      B.s[1] = *(const v8s*)&cB[(t * 16 + mr) * LDA + 16 + hi * 8];
      acc[t] = __builtin_amdgcn_wmma_f32_16x16x32_bf16(
          false, A.v, false, B.v, (short)0, acc[t], false, false);
    }
    __syncthreads();
  }

  // ---------------- epilogue ----------------
  const i64 base = (i64)blockIdx.x * TILE_M;
  #pragma unroll
  for (int r = 0; r < 8; ++r) {
    int row = m0 + r + 8 * hi;
    i64 g = base + row;
    if (g >= M) continue;
    i64 d = scatOut ? scatIdx[g] : 0;
    #pragma unroll
    for (int t = 0; t < 8; ++t) {
      int n = t * 16 + mr;
      float v = acc[t][r] + sBias2[n];
      if (outF)    outF[g * 128 + n]  = v;
      if (outBf)   outBf[g * 128 + n] = f2bf(v);
      if (scatOut) atomicAdd(scatOut + d * 128 + n, v);   // global_atomic_add_f32
    }
  }
}

// ---------------------------- helper kernels ------------------------------

__global__ void k_fill(float* p, i64 n, float v) {
  i64 i = (i64)blockIdx.x * 256 + threadIdx.x;
  if (i < n) p[i] = v;
}

__global__ void k_cvt_bf(const float* __restrict__ x, unsigned short* __restrict__ y, i64 n) {
  i64 i = (i64)blockIdx.x * 256 + threadIdx.x;
  if (i < n) y[i] = f2bf(x[i]);
}

__global__ void k_cvt_u(const float* __restrict__ u, float* uF, unsigned short* uBf) {
  int i = threadIdx.x;
  uF[i] = u[i];
  uBf[i] = f2bf(u[i]);
}

// Wt[n*K + k] = bf16(W[k*128 + n])   (transpose + convert)
__global__ void k_wprep(const float* __restrict__ w, unsigned short* __restrict__ wt, int K) {
  i64 i = (i64)blockIdx.x * 256 + threadIdx.x;
  if (i < (i64)K * 128) {
    int n = (int)(i / K), k = (int)(i % K);
    wt[i] = f2bf(w[(i64)k * 128 + n]);
  }
}

__global__ void k_count(const i64* __restrict__ col, float* cnt, i64 E) {
  i64 i = (i64)blockIdx.x * 256 + threadIdx.x;
  if (i < E) atomicAdd(&cnt[col[i]], 1.f);
}

__global__ void k_divide(const float* __restrict__ agg, const float* __restrict__ cnt,
                         unsigned short* __restrict__ aggBf, i64 n) {
  i64 i = (i64)blockIdx.x * 256 + threadIdx.x;
  if (i < n) {
    float c = cnt[i >> 7];
    c = c > 1.f ? c : 1.f;
    aggBf[i] = f2bf(agg[i] / c);
  }
}

// column mean over N rows: one block per feature
__global__ void k_colmean(const float* __restrict__ xF, float* __restrict__ mean, i64 N) {
  __shared__ float red[256];
  int b = blockIdx.x;
  float s = 0.f;
  for (i64 i = threadIdx.x; i < N; i += 256) s += xF[i * 128 + b];
  red[threadIdx.x] = s;
  __syncthreads();
  for (int o = 128; o > 0; o >>= 1) {
    if (threadIdx.x < o) red[threadIdx.x] += red[threadIdx.x + o];
    __syncthreads();
  }
  if (threadIdx.x == 0) mean[b] = red[0] / (float)N;
}

// tiny global-vector MLP: u = mlp(concat(u, xmean)); 1 block, 128 threads
__global__ void k_gmlp(const float* __restrict__ xmean,
                       const float* __restrict__ w1, const float* __restrict__ b1,
                       const float* __restrict__ w2, const float* __restrict__ b2,
                       float* uF, unsigned short* uBf) {
  __shared__ float sinp[256];
  __shared__ float sh[128];
  int n = threadIdx.x;
  sinp[n] = uF[n];
  sinp[128 + n] = xmean[n];
  __syncthreads();
  float s = b1[n];
  for (int k = 0; k < 256; ++k) s = fmaf(sinp[k], w1[k * 128 + n], s);
  sh[n] = s > 0.f ? s : 0.f;
  __syncthreads();
  float o = b2[n];
  for (int k = 0; k < 128; ++k) o = fmaf(sh[k], w2[k * 128 + n], o);
  __syncthreads();
  uF[n] = o;
  uBf[n] = f2bf(o);
}

__global__ void k_lin(const float* __restrict__ uF, const float* __restrict__ w,
                      const float* __restrict__ b, float* __restrict__ out) {
  int n = threadIdx.x;
  float o = b[n];
  for (int k = 0; k < 128; ++k) o = fmaf(uF[k], w[k * 128 + n], o);
  out[n] = o;
}

// ------------------------------- host side --------------------------------

extern "C" void kernel_launch(void* const* d_in, const int* in_sizes, int n_in,
                              void* d_out, int out_size, void* d_ws, size_t ws_size,
                              hipStream_t stream) {
  (void)n_in; (void)out_size; (void)ws_size;

  const float* x  = (const float*)d_in[0];
  const i64*   ei = (const i64*)d_in[1];        // edge_index int64 [2,E]
  const float* ea = (const float*)d_in[2];
  const float* u0 = (const float*)d_in[3];
  const i64 N = in_sizes[0] / 128;
  const i64 E = in_sizes[1] / 2;
  const i64* row = ei;
  const i64* col = ei + E;

  // params flattened in dict order: e1,n1a,n1b,g1,e2,n2a,n2b,g2,e3,n3a,n3b,g3
  // each MLP = l1.w,l1.b,l2.w,l2.b; then lin.w, lin.b
  auto P  = [&](int mlp, int t) -> const float* { return (const float*)d_in[4 + mlp * 4 + t]; };
  auto PK = [&](int mlp) -> int { return in_sizes[4 + mlp * 4] / 128; };

  // ---- workspace carve (all staged tensors must sit below 2^31 bytes) ----
  char* wsp = (char*)d_ws;
  auto alloc = [&](size_t bytes) -> char* {
    char* p = wsp;
    wsp += (bytes + 255) & ~(size_t)255;
    return p;
  };
  const char* wsBase = (const char*)d_ws;
  auto OFF = [&](const void* p) -> unsigned {
    return (unsigned)((const char*)p - wsBase);
  };

  unsigned short* x_bf   = (unsigned short*)alloc((size_t)N * 128 * 2);
  unsigned short* ea_bf  = (unsigned short*)alloc((size_t)E * 128 * 2);
  unsigned short* e_bf   = (unsigned short*)alloc((size_t)E * 128 * 2);
  float*          agg    = (float*)alloc((size_t)N * 128 * 4);
  unsigned short* agg_bf = (unsigned short*)alloc((size_t)N * 128 * 2);
  float*          cnt    = (float*)alloc((size_t)N * 4);
  float*          xF     = (float*)alloc((size_t)N * 128 * 4);
  float*          uF     = (float*)alloc(128 * 4);
  unsigned short* u_bf   = (unsigned short*)alloc(128 * 2);
  float*          xmean  = (float*)alloc(128 * 4);

  unsigned short* Wt1[12] = {};
  unsigned short* Wt2[12] = {};
  const int bigs[9] = {0, 1, 2, 4, 5, 6, 8, 9, 10};   // e1,n1a,n1b,e2,n2a,n2b,e3,n3a,n3b
  for (int i = 0; i < 9; ++i) {
    int m = bigs[i];
    int K = PK(m);
    Wt1[m] = (unsigned short*)alloc((size_t)K * 128 * 2);
    Wt2[m] = (unsigned short*)alloc((size_t)128 * 128 * 2);
  }

  // ---- weight prep (transpose + bf16) and input conversion ----
  for (int i = 0; i < 9; ++i) {
    int m = bigs[i];
    int K = PK(m);
    i64 n1 = (i64)K * 128;
    k_wprep<<<(unsigned)((n1 + 255) / 256), 256, 0, stream>>>(P(m, 0), Wt1[m], K);
    k_wprep<<<(unsigned)((16384 + 255) / 256), 256, 0, stream>>>(P(m, 2), Wt2[m], 128);
  }
  k_cvt_bf<<<(unsigned)((N * 128 + 255) / 256), 256, 0, stream>>>(x, x_bf, N * 128);
  k_cvt_bf<<<(unsigned)((E * 128 + 255) / 256), 256, 0, stream>>>(ea, ea_bf, E * 128);
  k_cvt_u<<<1, 128, 0, stream>>>(u0, uF, u_bf);

  const unsigned gE = (unsigned)((E + TILE_M - 1) / TILE_M);
  const unsigned gN = (unsigned)((N + TILE_M - 1) / TILE_M);

  // dispatch to the compile-time-K instantiations
  auto runMLP = [&](int K, unsigned grid, i64 M, int mlp,
                    unsigned a0, unsigned a1, unsigned a2, unsigned a3, int modes,
                    float* oF, unsigned short* oBf,
                    const i64* sIdx, float* sOut) {
    const unsigned w1o = OFF(Wt1[mlp]);
    const unsigned w2o = OFF(Wt2[mlp]);
    const float* bb1 = P(mlp, 1);
    const float* bb2 = P(mlp, 3);
    switch (K) {
      case 256:
        fused_mlp2<256><<<grid, 256, 0, stream>>>(M, wsBase, w1o, bb1, w2o, bb2,
            a0, a1, a2, a3, modes, row, col, oF, oBf, sIdx, sOut);
        break;
      case 384:
        fused_mlp2<384><<<grid, 256, 0, stream>>>(M, wsBase, w1o, bb1, w2o, bb2,
            a0, a1, a2, a3, modes, row, col, oF, oBf, sIdx, sOut);
        break;
      default:
        fused_mlp2<512><<<grid, 256, 0, stream>>>(M, wsBase, w1o, bb1, w2o, bb2,
            a0, a1, a2, a3, modes, row, col, oF, oBf, sIdx, sOut);
        break;
    }
  };

  struct Conv { int e, na, nb, g; int Ke, Knb; bool hasU; };
  const Conv convs[3] = {
      {0, 1, 2, 3, 384, 256, false},
      {4, 5, 6, 7, 512, 384, true},
      {8, 9, 10, 11, 512, 384, true},
  };

  for (int c = 0; c < 3; ++c) {
    const Conv& cv = convs[c];

    // edge MLP: e = mlp([x[row], x[col], e_prev(, u)]) -> e_bf (in-place safe)
    if (!cv.hasU) {
      runMLP(cv.Ke, gE, E, cv.e,
             OFF(x_bf), OFF(x_bf), OFF(ea_bf), 0u, 0x210,
             nullptr, e_bf, nullptr, nullptr);
    } else {
      runMLP(cv.Ke, gE, E, cv.e,
             OFF(x_bf), OFF(x_bf), OFF(e_bf), OFF(u_bf), 0x3210,
             nullptr, e_bf, nullptr, nullptr);
    }

    // scatter_mean(mlp([x[row], e]), col)
    k_fill<<<(unsigned)((N * 128 + 255) / 256), 256, 0, stream>>>(agg, N * 128, 0.f);
    k_fill<<<(unsigned)((N + 255) / 256), 256, 0, stream>>>(cnt, N, 0.f);
    runMLP(256, gE, E, cv.na,
           OFF(x_bf), OFF(e_bf), 0u, 0u, 0x20,
           nullptr, nullptr, col, agg);
    k_count<<<(unsigned)((E + 255) / 256), 256, 0, stream>>>(col, cnt, E);
    k_divide<<<(unsigned)((N * 128 + 255) / 256), 256, 0, stream>>>(agg, cnt, agg_bf, N * 128);

    // node MLP: x = mlp([x, agg(, u)]) -> xF (f32 for mean) + x_bf (in-place)
    if (!cv.hasU) {
      runMLP(cv.Knb, gN, N, cv.nb,
             OFF(x_bf), OFF(agg_bf), 0u, 0u, 0x22,
             xF, x_bf, nullptr, nullptr);
    } else {
      runMLP(cv.Knb, gN, N, cv.nb,
             OFF(x_bf), OFF(agg_bf), OFF(u_bf), 0u, 0x322,
             xF, x_bf, nullptr, nullptr);
    }

    // global MLP: u = mlp([u, mean(x)])
    k_colmean<<<128, 256, 0, stream>>>(xF, xmean, N);
    k_gmlp<<<1, 128, 0, stream>>>(xmean, P(cv.g, 0), P(cv.g, 1), P(cv.g, 2), P(cv.g, 3),
                                  uF, u_bf);
  }

  // final: out = u @ lin.w + lin.b
  k_lin<<<1, 128, 0, stream>>>(uF, (const float*)d_in[52], (const float*)d_in[53],
                               (float*)d_out);
}
